// mam3D_80745385164822
// MI455X (gfx1250) — matrix operand order repaired and checked by reading, not compiled, not run on hardware
//
#include <hip/hip_runtime.h>
#include <hip/hip_bf16.h>
#include <math.h>
#include <stdint.h>

// ---------------------------------------------------------------------------
// Model dims (from reference)
// ---------------------------------------------------------------------------
#define MB      4
#define NPATCH  1024
#define SEQL    (NPATCH + 1)          // 1025
#define PDIM    768
#define MDIM    512
#define OUTD    512
#define DIN     (2 * MDIM)            // 1024
#define DSTATE  16
#define DTRANK  (MDIM / 16)           // 32
#define KCONV   4
#define NBLK    2

typedef __attribute__((ext_vector_type(16))) __bf16 v16bf;
typedef __attribute__((ext_vector_type(8)))  __bf16 v8bf;
typedef __attribute__((ext_vector_type(8)))  float  v8f;

// ---------------------------------------------------------------------------
// CDNA5 async global->LDS copy (ASYNCcnt) with synchronous fallback.
// Builtin signature (from hipcc diagnostic): (v4i AS1*, v4i AS3*, Ii, Ii)
// ---------------------------------------------------------------------------
#if __has_builtin(__builtin_amdgcn_global_load_async_to_lds_b128) && \
    __has_builtin(__builtin_amdgcn_s_wait_asynccnt)
#define MAM_ASYNC_LDS 1
typedef int mam_v4i __attribute__((__vector_size__(4 * sizeof(int))));
typedef __attribute__((address_space(1))) mam_v4i* mam_gp1;
typedef __attribute__((address_space(3))) mam_v4i* mam_lp3;
__device__ __forceinline__ void mam_copy16(const __bf16* g, __bf16* l) {
    __builtin_amdgcn_global_load_async_to_lds_b128(
        (mam_gp1)(uintptr_t)g,
        (mam_lp3)(unsigned)(uintptr_t)l,
        0, 0);
}
#define MAM_ASYNC_WAIT() __builtin_amdgcn_s_wait_asynccnt(0)
#else
#define MAM_ASYNC_LDS 0
__device__ __forceinline__ void mam_copy16(const __bf16* g, __bf16* l) {
    *(v8bf*)l = *(const v8bf*)g;
}
#define MAM_ASYNC_WAIT() ((void)0)
#endif

// ---------------------------------------------------------------------------
// Helpers
// ---------------------------------------------------------------------------
__device__ __forceinline__ __bf16 f2bf(float f) {
    union { float f; unsigned u; } v; v.f = f;
    unsigned r = v.u + 0x7FFFu + ((v.u >> 16) & 1u);   // round-to-nearest-even
    unsigned short h = (unsigned short)(r >> 16);
    union { unsigned short s; __bf16 b; } o; o.s = h;
    return o.b;
}
__device__ __forceinline__ float siluf(float x) { return x / (1.f + __expf(-x)); }
__device__ __forceinline__ float softplusf(float x) {
    return (x > 20.f) ? x : log1pf(__expf(x));
}
__device__ __forceinline__ float geluf(float x) {
    return 0.5f * x * (1.f + erff(x * 0.70710678118654752f));
}
__device__ __forceinline__ float wave_sum(float v) {
#pragma unroll
    for (int o = 16; o > 0; o >>= 1) v += __shfl_down(v, o, 32);
    return v;
}

// ---------------------------------------------------------------------------
// fp32 -> bf16 bulk convert (row-major A operands)
// ---------------------------------------------------------------------------
__global__ __launch_bounds__(256)
void mam_f2bf_kernel(const float* __restrict__ in, __bf16* __restrict__ out, size_t n)
{
    const size_t i = ((size_t)blockIdx.x * 256 + threadIdx.x) * 4;
    if (i + 3 < n) {
        const float4 v = *(const float4*)(in + i);
        __bf16* p = out + i;
        p[0] = f2bf(v.x); p[1] = f2bf(v.y); p[2] = f2bf(v.z); p[3] = f2bf(v.w);
    } else {
        for (size_t j = i; j < n; ++j) out[j] = f2bf(in[j]);
    }
}

// ---------------------------------------------------------------------------
// Pack weights B[K][N] (f32, row-major) into WMMA-fragment-ready bf16 tiles:
//   out[((n16*(K/32) + ks)*32 + lane)*16 + j]
//   lane = r + 16*half holds col = n16*16 + r, k = ks*32 + 16*half + j
// ---------------------------------------------------------------------------
__global__ __launch_bounds__(256)
void mam_packB_kernel(const float* __restrict__ in, __bf16* __restrict__ out,
                      int K, int N)
{
    const size_t o = (size_t)blockIdx.x * 256 + threadIdx.x;
    if (o >= (size_t)K * N) return;
    const int j    = (int)(o & 15);
    const int lane = (int)((o >> 4) & 31);
    const size_t tks = o >> 9;
    const int K32 = K >> 5;
    const int ks  = (int)(tks % K32);
    const int n16 = (int)(tks / K32);
    const int col = n16 * 16 + (lane & 15);
    const int k   = ks * 32 + 16 * (lane >> 4) + j;
    out[o] = f2bf(in[(size_t)k * N + col]);
}

// ---------------------------------------------------------------------------
// bf16 WMMA GEMM: C[M,N](f32) = A[M,K](bf16 row-major) @ Bpk (packed bf16)
// 256 threads = 8 waves; block tile 64 x (16*NSUB); double-buffered LDS in
// fragment layout; async global->LDS staging when available.
// Wave (wm 0..3, wn 0..1): 16 x (8*NSUB) per wave -> NSUB/2 WMMA per slab.
// Requires K % 32 == 0, N % (16*NSUB) == 0; M tail via clamped rows.
// ---------------------------------------------------------------------------
#define EPI_NONE     0
#define EPI_SOFTPLUS 1
#define EPI_RESID    2

template <int OP, int NSUB>
__global__ __launch_bounds__(256)
void mam_gemm_bf16_wmma(const __bf16* __restrict__ A, const __bf16* __restrict__ Bpk,
                        const float* __restrict__ bias, const float* __restrict__ resid,
                        float* __restrict__ C, __bf16* __restrict__ Cbf,
                        int M, int N, int K, int lda, int ldc)
{
    __shared__ v16bf Af[2][4][32];       // [buf][M-subtile][lane]
    __shared__ v16bf Bf[2][NSUB][32];    // [buf][N-subtile][lane]

    const int tid  = threadIdx.x;
    const int lane = tid & 31;
    const int wave = tid >> 5;
    const int wm   = wave >> 1;                  // 0..3
    const int wn   = wave & 1;                   // 0..1
    const int half = lane >> 4;
    const int r    = lane & 15;

    const int blockM = blockIdx.y * 64;
    const int blockN = blockIdx.x * (16 * NSUB);
    const int K32 = K >> 5;
    const int nt0 = blockN >> 4;

    // ---- A staging: thread -> (row 0..63, k-octet 0..3), permuted dest ----
    const int arow  = tid >> 2;
    const int ag    = tid & 3;
    const int ahalf = ag >> 1;
    const int ajb   = (ag & 1) * 8;
    const int asub  = arow >> 4;
    const int alane = (arow & 15) + 16 * ahalf;
    const int grow  = blockM + arow;
    const int growc = (grow < M) ? grow : (M - 1);   // clamp: rows >= M never stored

    auto stageA = [&](int buf, int k0) {
        mam_copy16(A + (size_t)growc * lda + (k0 + ag * 8),
                   (__bf16*)&Af[buf][asub][alane] + ajb);
    };
    // ---- B staging: linear copy of packed tiles ---------------------------
    auto stageB = [&](int buf, int ks) {
#pragma unroll
        for (int q = 0; q < NSUB / 4; ++q) {
            const int c   = tid + q * 256;           // 0 .. NSUB*64-1
            const int tn  = c >> 6;
            const int rem = c & 63;
            mam_copy16(Bpk + (((size_t)(nt0 + tn) * K32 + ks) << 9) + rem * 8,
                       (__bf16*)&Bf[buf][tn][0] + rem * 8);
        }
    };

    v8f acc[NSUB / 2] = {};
    stageA(0, 0);
    stageB(0, 0);

    int buf = 0;
    for (int k0 = 0; k0 < K; k0 += 32, buf ^= 1) {
        MAM_ASYNC_WAIT();
        __syncthreads();
        if (k0 + 32 < K) {
            __builtin_prefetch(A + (size_t)growc * lda + (k0 + 64 + ag * 8), 0, 1);
            stageA(buf ^ 1, k0 + 32);
            stageB(buf ^ 1, (k0 >> 5) + 1);
        }
        const v16bf af = Af[buf][wm][lane];
#pragma unroll
        for (int t = 0; t < NSUB / 2; ++t) {
            const v16bf bt = Bf[buf][wn * (NSUB / 2) + t][lane];
            acc[t] = __builtin_amdgcn_wmma_f32_16x16x32_bf16(false, af, false, bt,
                                                             (short)0, acc[t],
                                                             false, false);
        }
    }

    // Epilogue: C/D layout -> VGPR v: row = v + 8*half, col = lane&15
    const int rowbase = blockM + wm * 16 + 8 * half;
#pragma unroll
    for (int v = 0; v < 8; ++v) {
        const int row = rowbase + v;
        if (row >= M) continue;
#pragma unroll
        for (int t = 0; t < NSUB / 2; ++t) {
            const int col = blockN + (wn * (NSUB / 2) + t) * 16 + r;
            float val = acc[t][v];
            if (bias) val += bias[col];
            if (OP == EPI_SOFTPLUS) val = softplusf(val);
            if (OP == EPI_RESID)    val += resid[(size_t)row * ldc + col];
            C[(size_t)row * ldc + col] = val;
            if (Cbf) Cbf[(size_t)row * ldc + col] = f2bf(val);
        }
    }
}

// ---------------------------------------------------------------------------
// Embed: x[b,0,:] = cls + pos[0];  x[b,1+p,:] = patch_out[b,p,:] + pos[1+p]
// ---------------------------------------------------------------------------
__global__ __launch_bounds__(256)
void mam_embed_kernel(const float* __restrict__ patch_out, const float* __restrict__ cls,
                      const float* __restrict__ pos, float* __restrict__ x)
{
    const size_t idx = (size_t)blockIdx.x * 256 + threadIdx.x;
    const int c = (int)(idx % MDIM);
    const size_t bt = idx / MDIM;
    const int t = (int)(bt % SEQL);
    const size_t b = bt / SEQL;
    float v;
    if (t == 0) v = cls[c];
    else        v = patch_out[(b * NPATCH + (t - 1)) * (size_t)MDIM + c];
    x[idx] = v + pos[(size_t)t * MDIM + c];
}

// ---------------------------------------------------------------------------
// Row LayerNorm (+optional GELU); optional f32 and/or bf16 outputs.
// ---------------------------------------------------------------------------
__global__ __launch_bounds__(256)
void mam_ln_kernel(const float* __restrict__ x, const float* __restrict__ w,
                   const float* __restrict__ b, float* __restrict__ outf,
                   __bf16* __restrict__ outb, int width, int do_gelu)
{
    const int row = blockIdx.x;
    const float* xr = x + (size_t)row * width;
    const int tid = threadIdx.x, lane = tid & 31, wave = tid >> 5;
    __shared__ float rs[8], rss[8], stats[2];
    float s = 0.f, ss = 0.f;
    for (int i = tid; i < width; i += 256) { float v = xr[i]; s += v; ss += v * v; }
    s = wave_sum(s); ss = wave_sum(ss);
    if (lane == 0) { rs[wave] = s; rss[wave] = ss; }
    __syncthreads();
    if (tid == 0) {
        float S = 0.f, SS = 0.f;
#pragma unroll
        for (int i = 0; i < 8; ++i) { S += rs[i]; SS += rss[i]; }
        const float mean = S / width;
        const float var  = SS / width - mean * mean;
        stats[0] = mean; stats[1] = rsqrtf(var + 1e-5f);
    }
    __syncthreads();
    const float mean = stats[0], inv = stats[1];
    for (int i = tid; i < width; i += 256) {
        float v = (xr[i] - mean) * inv * w[i] + b[i];
        if (do_gelu) v = geluf(v);
        if (outf) outf[(size_t)row * width + i] = v;
        if (outb) outb[(size_t)row * width + i] = f2bf(v);
    }
}

// ---------------------------------------------------------------------------
// Causal depthwise conv (K=4) + SiLU; outputs f32 (scan) and bf16 (GEMM A)
// ---------------------------------------------------------------------------
__global__ __launch_bounds__(256)
void mam_conv_silu_kernel(const float* __restrict__ xz, const float* __restrict__ cw,
                          const float* __restrict__ cb, float* __restrict__ u,
                          __bf16* __restrict__ ubf)
{
    const size_t idx = (size_t)blockIdx.x * 256 + threadIdx.x;
    const int d = (int)(idx & (DIN - 1));
    const size_t bt = idx >> 10;
    const int t = (int)(bt % SEQL);
    float acc = cb[d];
#pragma unroll
    for (int k = 0; k < KCONV; ++k) {
        const int tt = t - (KCONV - 1) + k;
        if (tt >= 0)
            acc += cw[d * KCONV + k] * xz[(bt - (KCONV - 1) + k) * (size_t)(2 * DIN) + d];
    }
    const float v = siluf(acc);
    u[idx]   = v;
    ubf[idx] = f2bf(v);
}

// ---------------------------------------------------------------------------
// Selective scan: one lane per (b,d); 16-state recurrence in VGPRs,
// B/C staged in LDS in chunks of 8 timesteps.
// ---------------------------------------------------------------------------
__global__ __launch_bounds__(256)
void mam_scan_kernel(const float* __restrict__ delta, const float* __restrict__ xdbl,
                     const float* __restrict__ u, const float* __restrict__ A_log,
                     const float* __restrict__ Dp, float* __restrict__ ycore)
{
    const int gid = blockIdx.x * 256 + threadIdx.x;
    const int b = gid >> 10;
    const int d = gid & (DIN - 1);
    const int tid = threadIdx.x;

    float a[DSTATE], h[DSTATE];
#pragma unroll
    for (int n = 0; n < DSTATE; ++n) { a[n] = -__expf(A_log[d * DSTATE + n]); h[n] = 0.f; }
    const float Dd = Dp[d];

    __shared__ float BC[8][32];
    const size_t baseBL = (size_t)b * SEQL;

    for (int t0 = 0; t0 < SEQL; t0 += 8) {
        __syncthreads();
        {
            const int tt = tid >> 5, cc = tid & 31;
            const int t = t0 + tt;
            BC[tt][cc] = (t < SEQL)
                ? xdbl[(baseBL + t) * (size_t)(DTRANK + 2 * DSTATE) + DTRANK + cc] : 0.f;
        }
        __syncthreads();
        const int tmax = (SEQL - t0 < 8) ? (SEQL - t0) : 8;
        for (int ti = 0; ti < tmax; ++ti) {
            const size_t row = baseBL + t0 + ti;
            const float dl = delta[row * DIN + d];
            const float ut = u[row * DIN + d];
            float y = 0.f;
#pragma unroll
            for (int n = 0; n < DSTATE; ++n) {
                const float dA  = __expf(dl * a[n]);
                const float dBu = dl * BC[ti][n] * ut;
                h[n] = dA * h[n] + dBu;
                y += h[n] * BC[ti][DSTATE + n];
            }
            ycore[row * DIN + d] = y + ut * Dd;
        }
    }
}

// ---------------------------------------------------------------------------
// Gate: ygated(bf16) = ycore * silu(z)
// ---------------------------------------------------------------------------
__global__ __launch_bounds__(256)
void mam_gate_kernel(const float* __restrict__ ycore, const float* __restrict__ xz,
                     __bf16* __restrict__ ygbf)
{
    const size_t idx = (size_t)blockIdx.x * 256 + threadIdx.x;
    const int d = (int)(idx & (DIN - 1));
    const size_t bt = idx >> 10;
    const float z = xz[bt * (size_t)(2 * DIN) + DIN + d];
    ygbf[idx] = f2bf(ycore[idx] * siluf(z));
}

// ---------------------------------------------------------------------------
// Softmax over sequence (per (b, channel)) + weighted pool of y.
// ---------------------------------------------------------------------------
__global__ __launch_bounds__(256)
void mam_pool_kernel(const float* __restrict__ y, const float* __restrict__ a,
                     float* __restrict__ pooled)
{
    const int b = blockIdx.y;
    const int j = blockIdx.x * 256 + threadIdx.x;
    const float* yb = y + (size_t)b * SEQL * OUTD;
    const float* ab = a + (size_t)b * SEQL * OUTD;
    float mx = -1e30f;
    for (int t = 0; t < SEQL; ++t) mx = fmaxf(mx, ab[(size_t)t * OUTD + j]);
    float se = 0.f, ws = 0.f;
    for (int t = 0; t < SEQL; ++t) {
        const float e = __expf(ab[(size_t)t * OUTD + j] - mx);
        se += e;
        ws += e * yb[(size_t)t * OUTD + j];
    }
    pooled[(size_t)b * OUTD + j] = ws / se;
}

// ---------------------------------------------------------------------------
// Host-side launcher
// ---------------------------------------------------------------------------
static inline void launch_gemm(int op, const __bf16* A, const __bf16* Bpk,
                               const float* bias, const float* resid,
                               float* C, __bf16* Cbf, int M, int N, int K,
                               int lda, int ldc, hipStream_t stream)
{
    if (N % 128 == 0) {
        dim3 g(N / 128, (M + 63) / 64);
        if (op == EPI_SOFTPLUS)
            mam_gemm_bf16_wmma<EPI_SOFTPLUS, 8><<<g, 256, 0, stream>>>(A, Bpk, bias, resid, C, Cbf, M, N, K, lda, ldc);
        else if (op == EPI_RESID)
            mam_gemm_bf16_wmma<EPI_RESID, 8><<<g, 256, 0, stream>>>(A, Bpk, bias, resid, C, Cbf, M, N, K, lda, ldc);
        else
            mam_gemm_bf16_wmma<EPI_NONE, 8><<<g, 256, 0, stream>>>(A, Bpk, bias, resid, C, Cbf, M, N, K, lda, ldc);
    } else {                                    // N % 64 == 0 (xproj N=64)
        dim3 g(N / 64, (M + 63) / 64);
        mam_gemm_bf16_wmma<EPI_NONE, 4><<<g, 256, 0, stream>>>(A, Bpk, bias, resid, C, Cbf, M, N, K, lda, ldc);
    }
}

static inline void launch_f2bf(const float* in, __bf16* out, size_t n, hipStream_t stream)
{
    const size_t nq = (n + 3) / 4;
    mam_f2bf_kernel<<<(unsigned)((nq + 255) / 256), 256, 0, stream>>>(in, out, n);
}

static inline void launch_packB(const float* in, __bf16* out, int K, int N,
                                hipStream_t stream)
{
    const size_t n = (size_t)K * N;
    mam_packB_kernel<<<(unsigned)((n + 255) / 256), 256, 0, stream>>>(in, out, K, N);
}

extern "C" void kernel_launch(void* const* d_in, const int* in_sizes, int n_in,
                              void* d_out, int out_size, void* d_ws, size_t ws_size,
                              hipStream_t stream)
{
    const float* img      = (const float*)d_in[0];
    const float* patch_W  = (const float*)d_in[1];
    const float* patch_b  = (const float*)d_in[2];
    const float* pos_emb  = (const float*)d_in[3];
    const float* cls_tok  = (const float*)d_in[4];
    const float* blk_ln_w = (const float*)d_in[5];
    const float* blk_ln_b = (const float*)d_in[6];
    const float* blk_in_W = (const float*)d_in[7];
    const float* blk_cw   = (const float*)d_in[8];
    const float* blk_cb   = (const float*)d_in[9];
    const float* blk_xp_W = (const float*)d_in[10];
    const float* blk_dt_W = (const float*)d_in[11];
    const float* blk_dt_b = (const float*)d_in[12];
    const float* blk_Alog = (const float*)d_in[13];
    const float* blk_D    = (const float*)d_in[14];
    const float* blk_o_W  = (const float*)d_in[15];
    const float* sl1_W    = (const float*)d_in[16];
    const float* sl1_b    = (const float*)d_in[17];
    const float* sl1_lnw  = (const float*)d_in[18];
    const float* sl1_lnb  = (const float*)d_in[19];
    const float* sl2_W    = (const float*)d_in[20];
    const float* sl2_b    = (const float*)d_in[21];
    const float* sl2_lnw  = (const float*)d_in[22];
    const float* sl2_lnb  = (const float*)d_in[23];
    const float* lat_lnw  = (const float*)d_in[24];
    const float* lat_lnb  = (const float*)d_in[25];
    float* out = (float*)d_out;

    const int ROWS = MB * SEQL;                 // 4100
    const int XPN  = DTRANK + 2 * DSTATE;       // 64

    // ---- byte-based workspace carve-out ----
    char* wsb = (char*)d_ws;
    size_t off = 0;
    auto allocf = [&](size_t n) { float* p = (float*)(wsb + off);
                                  off += ((n * 4 + 255) & ~(size_t)255); return p; };
    auto allocb = [&](size_t n) { __bf16* p = (__bf16*)(wsb + off);
                                  off += ((n * 2 + 255) & ~(size_t)255); return p; };
    // f32 activations
    float* patch_out = allocf((size_t)MB * NPATCH * MDIM);
    float* xbuf      = allocf((size_t)ROWS * MDIM);
    float* xz        = allocf((size_t)ROWS * 2 * DIN);
    float* ubuf      = allocf((size_t)ROWS * DIN);
    float* xdbl      = allocf((size_t)ROWS * XPN);
    float* deltab    = allocf((size_t)ROWS * DIN);
    float* ycore     = allocf((size_t)ROWS * DIN);
    float* hpre      = allocf((size_t)ROWS * OUTD);
    float* ybuf      = allocf((size_t)ROWS * OUTD);
    float* abuf      = allocf((size_t)ROWS * OUTD);
    float* pooled    = allocf((size_t)MB * OUTD);
    // bf16 row-major A operands
    __bf16* imgBF    = allocb((size_t)MB * NPATCH * PDIM);
    __bf16* xnBF     = allocb((size_t)ROWS * MDIM);
    __bf16* uBF      = allocb((size_t)ROWS * DIN);
    __bf16* xdblBF   = allocb((size_t)ROWS * XPN);
    __bf16* ygBF     = allocb((size_t)ROWS * DIN);
    __bf16* xbufBF   = allocb((size_t)ROWS * MDIM);
    __bf16* ybufBF   = allocb((size_t)ROWS * OUTD);
    // packed bf16 weights (fragment-ready B)
    __bf16* patchW_P = allocb((size_t)PDIM * MDIM);
    __bf16* inW_P    = allocb((size_t)NBLK * MDIM * 2 * DIN);
    __bf16* xpW_P    = allocb((size_t)NBLK * DIN * XPN);
    __bf16* dtW_P    = allocb((size_t)NBLK * DTRANK * DIN);
    __bf16* oW_P     = allocb((size_t)NBLK * DIN * MDIM);
    __bf16* sl1W_P   = allocb((size_t)OUTD * OUTD);
    __bf16* sl2W_P   = allocb((size_t)OUTD * OUTD);
    (void)ws_size; (void)n_in; (void)in_sizes; (void)out_size;

    // ---- conversions / weight packing ----
    launch_f2bf(img, imgBF, (size_t)MB * NPATCH * PDIM, stream);
    launch_packB(patch_W, patchW_P, PDIM, MDIM, stream);
    for (int i = 0; i < NBLK; ++i) {
        launch_packB(blk_in_W + (size_t)i * MDIM * 2 * DIN,
                     inW_P + (size_t)i * MDIM * 2 * DIN, MDIM, 2 * DIN, stream);
        launch_packB(blk_xp_W + (size_t)i * DIN * XPN,
                     xpW_P + (size_t)i * DIN * XPN, DIN, XPN, stream);
        launch_packB(blk_dt_W + (size_t)i * DTRANK * DIN,
                     dtW_P + (size_t)i * DTRANK * DIN, DTRANK, DIN, stream);
        launch_packB(blk_o_W + (size_t)i * DIN * MDIM,
                     oW_P + (size_t)i * DIN * MDIM, DIN, MDIM, stream);
    }
    launch_packB(sl1_W, sl1W_P, OUTD, OUTD, stream);
    launch_packB(sl2_W, sl2W_P, OUTD, OUTD, stream);

    // 1) patch embed
    launch_gemm(EPI_NONE, imgBF, patchW_P, patch_b, nullptr, patch_out, nullptr,
                MB * NPATCH, MDIM, PDIM, PDIM, MDIM, stream);
    // 2) cls concat + pos add
    mam_embed_kernel<<<(ROWS * MDIM) / 256, 256, 0, stream>>>(patch_out, cls_tok, pos_emb, xbuf);

    // 3) Mamba blocks
    for (int i = 0; i < NBLK; ++i) {
        const __bf16* in_W = inW_P + (size_t)i * MDIM * 2 * DIN;
        const __bf16* xp_W = xpW_P + (size_t)i * DIN * XPN;
        const __bf16* dt_W = dtW_P + (size_t)i * DTRANK * DIN;
        const float*  dt_b = blk_dt_b + (size_t)i * DIN;
        const float*  Alog = blk_Alog + (size_t)i * DIN * DSTATE;
        const float*  Dp   = blk_D    + (size_t)i * DIN;
        const __bf16* o_W  = oW_P + (size_t)i * DIN * MDIM;

        mam_ln_kernel<<<ROWS, 256, 0, stream>>>(xbuf, blk_ln_w + (size_t)i * MDIM,
                                                blk_ln_b + (size_t)i * MDIM,
                                                nullptr, xnBF, MDIM, 0);
        launch_gemm(EPI_NONE, xnBF, in_W, nullptr, nullptr, xz, nullptr,
                    ROWS, 2 * DIN, MDIM, MDIM, 2 * DIN, stream);
        mam_conv_silu_kernel<<<(ROWS * DIN) / 256, 256, 0, stream>>>(
            xz, blk_cw + (size_t)i * DIN * KCONV, blk_cb + (size_t)i * DIN, ubuf, uBF);
        launch_gemm(EPI_NONE, uBF, xp_W, nullptr, nullptr, xdbl, xdblBF,
                    ROWS, XPN, DIN, DIN, XPN, stream);
        launch_gemm(EPI_SOFTPLUS, xdblBF, dt_W, dt_b, nullptr, deltab, nullptr,
                    ROWS, DIN, DTRANK, XPN, DIN, stream);
        mam_scan_kernel<<<(MB * DIN) / 256, 256, 0, stream>>>(deltab, xdbl, ubuf, Alog, Dp, ycore);
        mam_gate_kernel<<<(ROWS * DIN) / 256, 256, 0, stream>>>(ycore, xz, ygBF);
        launch_gemm(EPI_RESID, ygBF, o_W, nullptr, xbuf, xbuf, xbufBF,
                    ROWS, MDIM, DIN, DIN, MDIM, stream);
    }

    // 4) head
    launch_gemm(EPI_NONE, xbufBF, sl1W_P, sl1_b, nullptr, hpre, nullptr,
                ROWS, OUTD, MDIM, MDIM, OUTD, stream);
    mam_ln_kernel<<<ROWS, 256, 0, stream>>>(hpre, sl1_lnw, sl1_lnb, ybuf, ybufBF, OUTD, 1);
    launch_gemm(EPI_NONE, ybufBF, sl2W_P, sl2_b, nullptr, hpre, nullptr,
                ROWS, OUTD, OUTD, OUTD, OUTD, stream);
    mam_ln_kernel<<<ROWS, 256, 0, stream>>>(hpre, sl2_lnw, sl2_lnb, abuf, nullptr, OUTD, 1);
    {
        dim3 g(OUTD / 256, MB);
        mam_pool_kernel<<<g, 256, 0, stream>>>(ybuf, abuf, pooled);
    }
    mam_ln_kernel<<<MB, 256, 0, stream>>>(pooled, lat_lnw, lat_lnb, out, nullptr, OUTD, 0);
}
